// _EAxisCrossAttention_32899449488094
// MI455X (gfx1250) — matrix operand order repaired and checked by reading, not compiled, not run on hardware
//
#include <hip/hip_runtime.h>
#include <math.h>

namespace {
constexpr int kH  = 1024;
constexpr int kDK = 256;
constexpr int kTokPerWG = 8;                    // 64 rows (M=64) per workgroup
constexpr int kWGs = 4 * 2048 / kTokPerWG;      // 1024 workgroups
// dynamic LDS layout (bytes)
constexpr int kAdOff = 0;                        // 64 rows x 1024 bf16 = 131072
constexpr int kUlOff = kAdOff + 64 * kH * 2;     // 8 rows x 1024 bf16  =  16384
constexpr int kKlOff = kUlOff + 8 * kH * 2;      // 64 x 256 bf16       =  32768
constexpr int kQlOff = kKlOff + 64 * kDK * 2;    // 64 x 256 bf16       =  32768
constexpr int kSlOff = kQlOff + 64 * kDK * 2;    // 4 x 16 x 16 f32     =   4096
constexpr int kPwOff = kSlOff + 4 * 256 * 4;     // 64 x 8 f32          =   2048
constexpr int kSmemBytes = kPwOff + 64 * 8 * 4;  // 219136 total (<320KB)
}

typedef __bf16 bf16_t;
typedef __attribute__((ext_vector_type(16))) __bf16 v16bf;
typedef __attribute__((ext_vector_type(8)))  __bf16 v8bf;
typedef __attribute__((ext_vector_type(4)))  __bf16 v4bf;
typedef __attribute__((ext_vector_type(8)))  float  v8f;

// Build a 16-element bf16 fragment from two contiguous 16-byte chunks.
static __device__ __forceinline__ v16bf frag2(const bf16_t* p0, const bf16_t* p1) {
  union { v16bf v; uint4 q[2]; } u;
  u.q[0] = *(const uint4*)p0;
  u.q[1] = *(const uint4*)p1;
  return u.v;
}

static __device__ __forceinline__ v8f wmma_bf16(v16bf a, v16bf b, v8f c) {
  // D = A(16x32) * B(32x16) + C(16x16 f32)
  return __builtin_amdgcn_wmma_f32_16x16x32_bf16(false, a, false, b, (short)0, c,
                                                 false, false);
}

// ---------------- prep kernels ----------------

// WT[n*K + k] = (bf16) W[k*N + n]
__global__ void k_transpose_cast(const float* __restrict__ W, bf16_t* __restrict__ WT,
                                 int K, int N) {
  int idx = blockIdx.x * blockDim.x + threadIdx.x;
  if (idx >= K * N) return;
  int n = idx / K;
  int k = idx - n * K;
  WT[idx] = (bf16_t)W[k * N + n];
}

// WvoT[b*H + a] = (bf16) sum_j Wv[a][j] * Wo[j][b]   (f32 accumulation, LDS-tiled)
__global__ __launch_bounds__(256) void k_wvo(const float* __restrict__ Wv,
                                             const float* __restrict__ Wo,
                                             bf16_t* __restrict__ WvoT) {
  __shared__ float Av[64][65];
  __shared__ float Bo[64][65];
  const int a0 = (blockIdx.x & 15) * 64;
  const int b0 = (blockIdx.x >> 4) * 64;
  const int tid = threadIdx.x;
  const int tx = tid & 15, ty = tid >> 4;
  float acc[16] = {};
  for (int j0 = 0; j0 < kH; j0 += 64) {
    for (int t = tid; t < 64 * 64; t += 256) {
      int r = t >> 6, c = t & 63;
      Av[r][c] = Wv[(a0 + r) * kH + j0 + c];
      Bo[r][c] = Wo[(j0 + r) * kH + b0 + c];
    }
    __syncthreads();
#pragma unroll 8
    for (int j = 0; j < 64; ++j)
#pragma unroll
      for (int ii = 0; ii < 4; ++ii)
#pragma unroll
        for (int jj = 0; jj < 4; ++jj)
          acc[ii * 4 + jj] += Av[ty * 4 + ii][j] * Bo[j][tx * 4 + jj];
    __syncthreads();
  }
#pragma unroll
  for (int ii = 0; ii < 4; ++ii)
#pragma unroll
    for (int jj = 0; jj < 4; ++jj)
      WvoT[(b0 + tx * 4 + jj) * kH + (a0 + ty * 4 + ii)] = (bf16_t)acc[ii * 4 + jj];
}

// ---------------- fused attention kernel ----------------
// One workgroup = 8 tokens = 64 rows (four 16-row M-tiles). 8 waves. Final GEMM
// (M=64, N=1024, K=1024) done in two N-passes; per pass each wave owns a 64-col
// strip (4 N-tiles) and shares each B fragment across all 4 M-tiles.
__global__ __launch_bounds__(256) void k_attn(
    const float* __restrict__ u, const float* __restrict__ deltas,
    const float* __restrict__ q_bias,
    const bf16_t* __restrict__ WkT, const bf16_t* __restrict__ WqT,
    const bf16_t* __restrict__ WvoT, float* __restrict__ out) {
  extern __shared__ __align__(16) char smem[];
  bf16_t* Ad = (bf16_t*)(smem + kAdOff);   // deltas rows (bf16); later holds P in place
  bf16_t* Ul = (bf16_t*)(smem + kUlOff);   // u rows (bf16)
  bf16_t* Kl = (bf16_t*)(smem + kKlOff);   // K projection (bf16)
  bf16_t* Ql = (bf16_t*)(smem + kQlOff);   // Q projection (bf16)
  float*  Sl = (float*) (smem + kSlOff);   // raw scores (4 tiles of 16x16)
  float*  Pw = (float*) (smem + kPwOff);   // softmax weights (64 x 8)

  const int tid  = threadIdx.x;
  const int lane = tid & 31;
  const int wave = tid >> 5;
  const int m    = lane & 15;   // tile row / B column index per lane
  const int half = lane >> 4;

  const float* dsrc = deltas + (size_t)blockIdx.x * 64 * kH;
  float*       dst  = out    + (size_t)blockIdx.x * 64 * kH;
  const float* usrc = u      + (size_t)blockIdx.x * 8 * kH;

  // ---- phase 0: load + cast inputs into LDS (16B vectorized DS stores) ----
  for (int i = tid; i < 64 * kH / 8; i += 256) {
    float4 v0 = ((const float4*)dsrc)[2 * i];
    float4 v1 = ((const float4*)dsrc)[2 * i + 1];
    v8bf o;
    o[0] = (bf16_t)v0.x; o[1] = (bf16_t)v0.y; o[2] = (bf16_t)v0.z; o[3] = (bf16_t)v0.w;
    o[4] = (bf16_t)v1.x; o[5] = (bf16_t)v1.y; o[6] = (bf16_t)v1.z; o[7] = (bf16_t)v1.w;
    *(v8bf*)(Ad + i * 8) = o;
  }
  for (int i = tid; i < 8 * kH / 8; i += 256) {
    float4 v0 = ((const float4*)usrc)[2 * i];
    float4 v1 = ((const float4*)usrc)[2 * i + 1];
    v8bf o;
    o[0] = (bf16_t)v0.x; o[1] = (bf16_t)v0.y; o[2] = (bf16_t)v0.z; o[3] = (bf16_t)v0.w;
    o[4] = (bf16_t)v1.x; o[5] = (bf16_t)v1.y; o[6] = (bf16_t)v1.z; o[7] = (bf16_t)v1.w;
    *(v8bf*)(Ul + i * 8) = o;
  }
  __syncthreads();

  // ---- phase 1: K = Ad @ WkT, Q = rep(u) @ WqT + bias ----
  // Wave handles N-tiles 2w, 2w+1; each B fragment shared across all 4 M-tiles.
#pragma unroll
  for (int s = 0; s < 2; ++s) {
    const int n0 = (wave * 2 + s) * 16;
    v8f ck[4] = {}, cq[4] = {};
    const bf16_t* bkrow = WkT + (size_t)(n0 + m) * kH + half * 8;
    const bf16_t* bqrow = WqT + (size_t)(n0 + m) * kH + half * 8;
    for (int k0 = 0; k0 < kH; k0 += 32) {
      v16bf bk = frag2(bkrow + k0, bkrow + k0 + 16);
      v16bf bq = frag2(bqrow + k0, bqrow + k0 + 16);
#pragma unroll
      for (int mt = 0; mt < 4; ++mt) {
        const bf16_t* arow = Ad + (size_t)(mt * 16 + m) * kH + half * 8 + k0;
        v16bf a = frag2(arow, arow + 16);
        ck[mt] = wmma_bf16(a, bk, ck[mt]);
        const int tok = mt * 2 + (m >> 3);
        const bf16_t* qrow = Ul + (size_t)tok * kH + half * 8 + k0;
        v16bf aq = frag2(qrow, qrow + 16);
        cq[mt] = wmma_bf16(aq, bq, cq[mt]);
      }
    }
#pragma unroll
    for (int mt = 0; mt < 4; ++mt)
#pragma unroll
      for (int j = 0; j < 8; ++j) {   // D layout: col n=m, VGPR j -> row j+8*half; expert e == j
        const int row = mt * 16 + j + half * 8;
        Kl[row * kDK + n0 + m] = (bf16_t)ck[mt][j];
        Ql[row * kDK + n0 + m] = (bf16_t)(cq[mt][j] + q_bias[j * kDK + n0 + m]);
      }
  }
  __syncthreads();

  // ---- phase 2: S = Q @ K^T, four 16x16 tiles (waves 0..3), scale 1/sqrt(256) ----
  if (wave < 4) {
    const int mt = wave;
    v8f cs = {};
    const bf16_t* arow = Ql + (size_t)(mt * 16 + m) * kDK + half * 8;
    const bf16_t* brow = Kl + (size_t)(mt * 16 + m) * kDK + half * 8;  // B col n = K row n
#pragma unroll
    for (int k0 = 0; k0 < kDK; k0 += 32) {
      v16bf a = frag2(arow + k0, arow + k0 + 16);
      v16bf b = frag2(brow + k0, brow + k0 + 16);
      cs = wmma_bf16(a, b, cs);
    }
#pragma unroll
    for (int j = 0; j < 8; ++j) Sl[mt * 256 + (j + half * 8) * 16 + m] = cs[j] * 0.0625f;
  }
  __syncthreads();

  // ---- softmax over per-token diagonal 8x8 blocks (64 rows) ----
  if (tid < 64) {
    const int r = tid;
    const int mt = r >> 4;
    const int rr = r & 15;
    const int cb = ((r >> 3) & 1) * 8;     // token block base column within tile
    float sc[8], mx = -3.0e38f;
#pragma unroll
    for (int f = 0; f < 8; ++f) { sc[f] = Sl[mt * 256 + rr * 16 + cb + f]; mx = fmaxf(mx, sc[f]); }
    float sum = 0.f;
#pragma unroll
    for (int f = 0; f < 8; ++f) { sc[f] = __expf(sc[f] - mx); sum += sc[f]; }
    const float inv = 1.0f / sum;
#pragma unroll
    for (int f = 0; f < 8; ++f) Pw[r * 8 + f] = sc[f] * inv;
  }
  __syncthreads();

  // ---- phase P: Ad <- attn @ Ad, in place ----
  // Thread owns 4 consecutive columns (8B DS ops). Token groups are independent
  // (each 8x8 attention block reads/writes only its own 8 rows), and the 8 input
  // rows are buffered in registers before the in-place write.
  {
    const int h0 = tid * 4;
#pragma unroll 1
    for (int g = 0; g < 8; ++g) {
      float df[8][4];
#pragma unroll
      for (int f = 0; f < 8; ++f) {
        v4bf d = *(const v4bf*)(Ad + (size_t)(g * 8 + f) * kH + h0);
        df[f][0] = (float)d[0]; df[f][1] = (float)d[1];
        df[f][2] = (float)d[2]; df[f][3] = (float)d[3];
      }
      v4bf pv[8];
#pragma unroll
      for (int rr = 0; rr < 8; ++rr) {
        const float* w = Pw + (g * 8 + rr) * 8;
        float o0 = 0.f, o1 = 0.f, o2 = 0.f, o3 = 0.f;
#pragma unroll
        for (int f = 0; f < 8; ++f) {
          const float wf = w[f];
          o0 += wf * df[f][0]; o1 += wf * df[f][1];
          o2 += wf * df[f][2]; o3 += wf * df[f][3];
        }
        v4bf ov;
        ov[0] = (bf16_t)o0; ov[1] = (bf16_t)o1; ov[2] = (bf16_t)o2; ov[3] = (bf16_t)o3;
        pv[rr] = ov;
      }
#pragma unroll
      for (int rr = 0; rr < 8; ++rr)
        *(v4bf*)(Ad + (size_t)(g * 8 + rr) * kH + h0) = pv[rr];
    }
  }
  __syncthreads();

  // ---- phase 3: out = deltas + P @ WvoT, two N-passes ----
  // Per pass each wave owns 64 cols (4 N-tiles) x 4 M-tiles; each B fragment is
  // reused by all 4 M-tiles; the two passes touch disjoint halves of WvoT.
#pragma unroll 1
  for (int p = 0; p < 2; ++p) {
    v8f acc[4][4] = {};                    // [mt][nt], 128 VGPRs
    const int nb = p * 512 + wave * 64;
    const bf16_t* brow = WvoT + (size_t)(nb + m) * kH + half * 8;
    for (int k0 = 0; k0 < kH; k0 += 32) {
      v16bf b[4];
#pragma unroll
      for (int t = 0; t < 4; ++t) {
        const bf16_t* bp = brow + (size_t)t * 16 * kH + k0;
        b[t] = frag2(bp, bp + 16);
      }
      v16bf a[4];
#pragma unroll
      for (int mt = 0; mt < 4; ++mt) {
        const bf16_t* ap = Ad + (size_t)(mt * 16 + m) * kH + half * 8 + k0;
        a[mt] = frag2(ap, ap + 16);
      }
#pragma unroll
      for (int mt = 0; mt < 4; ++mt)
#pragma unroll
        for (int t = 0; t < 4; ++t)
          acc[mt][t] = wmma_bf16(a[mt], b[t], acc[mt][t]);
    }
#pragma unroll
    for (int mt = 0; mt < 4; ++mt)
#pragma unroll
      for (int t = 0; t < 4; ++t) {
        const int col = nb + t * 16 + m;
#pragma unroll
        for (int j = 0; j < 8; ++j) {
          const int row = mt * 16 + j + half * 8;
          dst[row * kH + col] = dsrc[row * kH + col] + acc[mt][t][j];
        }
      }
  }
}

// ---------------- host entry ----------------
extern "C" void kernel_launch(void* const* d_in, const int* in_sizes, int n_in,
                              void* d_out, int out_size, void* d_ws, size_t ws_size,
                              hipStream_t stream) {
  (void)in_sizes; (void)n_in; (void)out_size; (void)ws_size;
  const float* u      = (const float*)d_in[0];
  const float* deltas = (const float*)d_in[1];
  const float* Wq     = (const float*)d_in[2];
  const float* Wk     = (const float*)d_in[3];
  const float* Wv     = (const float*)d_in[4];
  const float* Wo     = (const float*)d_in[5];
  const float* qb     = (const float*)d_in[6];
  float* out = (float*)d_out;

  // workspace: bf16 transposed weights ([N][K] row-major) — 3 MB total
  bf16_t* WkT  = (bf16_t*)d_ws;            // 256*1024
  bf16_t* WqT  = WkT + kDK * kH;           // 256*1024
  bf16_t* WvoT = WqT + kDK * kH;           // 1024*1024

  // Opt in to >64KB dynamic LDS (CDNA5 allows up to 320KB per workgroup).
  (void)hipFuncSetAttribute((const void*)k_attn,
                            hipFuncAttributeMaxDynamicSharedMemorySize, kSmemBytes);

  k_transpose_cast<<<(kH * kDK + 255) / 256, 256, 0, stream>>>(Wk, WkT, kH, kDK);
  k_transpose_cast<<<(kH * kDK + 255) / 256, 256, 0, stream>>>(Wq, WqT, kH, kDK);
  k_wvo<<<256, 256, 0, stream>>>(Wv, Wo, WvoT);
  k_attn<<<kWGs, 256, kSmemBytes, stream>>>(u, deltas, qb, WkT, WqT, WvoT, out);
}
